// TransformerXLModel_53034256171184
// MI455X (gfx1250) — matrix operand order repaired and checked by reading, not compiled, not run on hardware
//
#include <hip/hip_runtime.h>
#include <hip/hip_bf16.h>
#include <stdint.h>
#include <stddef.h>

#define QLEN 512
#define MLEN 512
#define KLEN (QLEN + MLEN)   // 1024
#define RLEN (KLEN + QLEN)   // 1536
#define BATCH 8
#define NH 16
#define DH 64
#define DMODEL 1024
#define DI 4096
#define NTOKQ (QLEN * BATCH) // 4096
#define NTOKK (KLEN * BATCH) // 8192

typedef __bf16 bf16;
typedef __attribute__((ext_vector_type(16))) __bf16 v16bf;
typedef __attribute__((ext_vector_type(8)))  __bf16 v8bf;
typedef __attribute__((ext_vector_type(8)))  float  v8f;
typedef __attribute__((ext_vector_type(4))) unsigned int v4u;
typedef __attribute__((ext_vector_type(8))) int v8i;
typedef __attribute__((ext_vector_type(4))) int v4i;

#if defined(__has_builtin)
#if __has_builtin(__builtin_amdgcn_tensor_load_to_lds)
#define HAVE_TDM 1
#endif
#if __has_builtin(__builtin_amdgcn_s_wait_tensorcnt)
#define HAVE_WAIT_TENSOR 1
#endif
#endif

// ---------------- helpers ----------------

__device__ __forceinline__ bf16 f2bf(float f) {
    unsigned u = __builtin_bit_cast(unsigned, f);
    unsigned r = u + 0x7FFFu + ((u >> 16) & 1u);   // round-to-nearest-even
    unsigned short h = (unsigned short)(r >> 16);
    return __builtin_bit_cast(bf16, h);
}

__device__ __forceinline__ float geluf(float x) {
    float c = 0.7978845608028654f; // sqrt(2/pi)
    float t = tanhf(c * (x + 0.044715f * x * x * x));
    return x * 0.5f * (1.0f + t);
}

// A fragment: 16x32 bf16 (rows m0..m0+15, K k0..k0+31), ISA 7.12.2 layout:
//   lanes 0-15 : row = lane,    elems 0-7 -> K k0+0..7,  elems 8-15 -> K k0+16..23
//   lanes 16-31: row = lane-16, elems 0-7 -> K k0+8..15, elems 8-15 -> K k0+24..31
__device__ __forceinline__ v16bf load_a_frag(const bf16* A, int lda, int m0, int k0, int lane) {
    int r  = m0 + (lane & 15);
    int kb = k0 + ((lane & 16) ? 8 : 0);
    const bf16* p = A + (size_t)r * lda + kb;
    v8bf lo = *(const v8bf*)(p);
    v8bf hi = *(const v8bf*)(p + 16);
    v16bf o;
#pragma unroll
    for (int e = 0; e < 8; ++e) { o[e] = lo[e]; o[e + 8] = hi[e]; }
    return o;
}

// B fragment: 32x16 bf16 (K k0..k0+31, cols n0..n0+15) from Bt stored [N][K]:
//   lanes 0-15 : col = lane,    elems 0-7 -> K k0+0..7,  elems 8-15 -> K k0+16..23
//   lanes 16-31: col = lane-16, elems 0-7 -> K k0+8..15, elems 8-15 -> K k0+24..31
__device__ __forceinline__ v16bf load_b_frag(const bf16* Bt, int ldb, int n0, int k0, int lane) {
    int c  = n0 + (lane & 15);
    int kb = k0 + ((lane & 16) ? 8 : 0);
    const bf16* p = Bt + (size_t)c * ldb + kb;
    v8bf lo = *(const v8bf*)(p);
    v8bf hi = *(const v8bf*)(p + 16);
    v16bf o;
#pragma unroll
    for (int e = 0; e < 8; ++e) { o[e] = lo[e]; o[e + 8] = hi[e]; }
    return o;
}

__device__ __forceinline__ v8f wmma_bf16(v16bf a, v16bf b, v8f c) {
    return __builtin_amdgcn_wmma_f32_16x16x32_bf16(false, a, false, b, (short)0, c, false, false);
}

#ifdef HAVE_TDM
// TDM: DMA one 2-D bf16 tile (rows x rowElems, row stride strideElems) from
// global memory into LDS at byte offset lds_off. D# per CDNA5 ISA ch.8.
__device__ __forceinline__ void tdm_load_2d(unsigned lds_off, const void* gaddr,
                                            unsigned rowElems, unsigned rows,
                                            unsigned strideElems) {
    unsigned long long ga = (unsigned long long)(uintptr_t)gaddr;
    v4u g0;
    g0[0] = 1u;                                        // count=1, user descriptor
    g0[1] = lds_off;                                   // lds_addr (bytes)
    g0[2] = (unsigned)(ga & 0xffffffffu);              // global_addr[31:0]
    g0[3] = (unsigned)((ga >> 32) & 0x01ffffffu)       // global_addr[56:32]
            | 0x80000000u;                             // type=2 ("image")
    v8i g1;
    g1[0] = 0x00010000;                  // wg_mask=0, data_size=1 (2 bytes)
    g1[1] = (int)(rowElems << 16);       // tensor_dim0[15:0]  (bits 63:48)
    g1[2] = (int)(rows << 16);           // tensor_dim1[15:0]  (bits 95:80)
    g1[3] = (int)(rowElems << 16);       // tile_dim0          (bits 127:112)
    g1[4] = (int)rows;                   // tile_dim1          (bits 143:128)
    g1[5] = (int)strideElems;            // tensor_dim0_stride[31:0]
    g1[6] = 0;
    g1[7] = 0;
    v4i z4 = {0, 0, 0, 0};
#if __clang_major__ >= 23
    v8i z8 = {0, 0, 0, 0, 0, 0, 0, 0};
    __builtin_amdgcn_tensor_load_to_lds(g0, g1, z4, z4, z8, 0);
#else
    __builtin_amdgcn_tensor_load_to_lds(g0, g1, z4, z4, 0);
#endif
}

__device__ __forceinline__ void wait_tensorcnt0() {
#ifdef HAVE_WAIT_TENSOR
    __builtin_amdgcn_s_wait_tensorcnt(0);
#else
    asm volatile("s_wait_tensorcnt 0x0" ::: "memory");
#endif
}
#endif // HAVE_TDM

// ---------------- elementwise / setup kernels ----------------

__global__ __launch_bounds__(256) void convert_transpose_kernel(
        const float* __restrict__ W, bf16* __restrict__ Wt, int K, int N) {
    size_t total = (size_t)K * N;
    for (size_t idx = (size_t)blockIdx.x * 256 + threadIdx.x; idx < total;
         idx += (size_t)gridDim.x * 256) {
        size_t k = idx / N; size_t n = idx % N;
        Wt[n * (size_t)K + k] = f2bf(W[idx]);
    }
}

__global__ __launch_bounds__(256) void convert_copy_kernel(
        const float* __restrict__ W, bf16* __restrict__ O, size_t total) {
    for (size_t idx = (size_t)blockIdx.x * 256 + threadIdx.x; idx < total;
         idx += (size_t)gridDim.x * 256)
        O[idx] = f2bf(W[idx]);
}

// cat = [mems ; lookup(inp_ids)]  -> bf16 [KLEN*B][D]; also h (query part) as f32
__global__ __launch_bounds__(256) void embed_kernel(
        const int* __restrict__ ids, const float* __restrict__ mems,
        const float* __restrict__ lut, bf16* __restrict__ cat_b,
        float* __restrict__ h_f) {
    size_t total = (size_t)NTOKK * DMODEL;
    for (size_t idx = (size_t)blockIdx.x * 256 + threadIdx.x; idx < total;
         idx += (size_t)gridDim.x * 256) {
        size_t row = idx / DMODEL; int c = (int)(idx % DMODEL);
        float v;
        if (row < (size_t)MLEN * BATCH) {
            v = mems[idx];
        } else {
            size_t qrow = row - (size_t)MLEN * BATCH; // i*B + b
            int tok = ids[qrow];
            v = lut[(size_t)tok * DMODEL + c];
            h_f[qrow * DMODEL + c] = v;
        }
        cat_b[idx] = f2bf(v);
    }
}

// sinusoidal relative positional embedding, pos_seq[r] = KLEN - r
__global__ __launch_bounds__(256) void posemb_kernel(bf16* __restrict__ pos) {
    size_t total = (size_t)RLEN * DMODEL;
    for (size_t idx = (size_t)blockIdx.x * 256 + threadIdx.x; idx < total;
         idx += (size_t)gridDim.x * 256) {
        int r = (int)(idx / DMODEL), c = (int)(idx % DMODEL);
        float p = (float)(KLEN - r);
        int t = (c < DMODEL / 2) ? c : c - DMODEL / 2;
        float f = __powf(10000.0f, -(float)t / (DMODEL / 2));
        float a = p * f;
        pos[idx] = f2bf((c < DMODEL / 2) ? __sinf(a) : __cosf(a));
    }
}

__global__ __launch_bounds__(256) void layernorm_kernel(
        const float* __restrict__ in, const float* __restrict__ g,
        const float* __restrict__ be, float* __restrict__ outf,
        bf16* __restrict__ outb) {
    int row = blockIdx.x;
    const float* x = in + (size_t)row * DMODEL;
    int tid = threadIdx.x, w = tid >> 5, lane = tid & 31;
    float s = 0.f, s2 = 0.f;
    for (int c = tid; c < DMODEL; c += 256) { float v = x[c]; s += v; s2 += v * v; }
#pragma unroll
    for (int off = 16; off; off >>= 1) { s += __shfl_xor(s, off, 32); s2 += __shfl_xor(s2, off, 32); }
    __shared__ float red[2][8];
    if (lane == 0) { red[0][w] = s; red[1][w] = s2; }
    __syncthreads();
    if (w == 0) {
        s  = (lane < 8) ? red[0][lane] : 0.f;
        s2 = (lane < 8) ? red[1][lane] : 0.f;
#pragma unroll
        for (int off = 4; off; off >>= 1) { s += __shfl_xor(s, off, 32); s2 += __shfl_xor(s2, off, 32); }
        if (lane == 0) { red[0][0] = s; red[1][0] = s2; }
    }
    __syncthreads();
    float mean = red[0][0] * (1.0f / DMODEL);
    float var  = red[1][0] * (1.0f / DMODEL) - mean * mean;
    float rstd = rsqrtf(var + 1e-12f);
    for (int c = tid; c < DMODEL; c += 256) {
        float v = (x[c] - mean) * rstd * g[c] + be[c];
        outf[(size_t)row * DMODEL + c] = v;
        if (outb) outb[(size_t)row * DMODEL + c] = f2bf(v);
    }
}

// ---------------- generic bf16 WMMA GEMM with fused epilogues ----------------
// C[M][N] = sum_k A[m][k] * Bt[n][k]   (A row-major [M][K], Bt row-major [N][K])
// 256 thr = 8 waves; block tile 128x128; wave tile 64x32 (4x2 frags)
#define EPI_STORE_BF16 0
#define EPI_QPROJ      1
#define EPI_KPROJ      2
#define EPI_VPROJ      3
#define EPI_RPROJ      4
#define EPI_OPROJ      5
#define EPI_FF1        6
#define EPI_FF2        7

__global__ __launch_bounds__(256) void gemm_bf16_wmma_kernel(
        const bf16* __restrict__ A, const bf16* __restrict__ Bt,
        int M, int N, int K, int mode,
        const float* __restrict__ p0, const float* __restrict__ p1,
        const float* __restrict__ p2,
        void* __restrict__ out0, void* __restrict__ out1) {
    const int tid = threadIdx.x, w = tid >> 5, lane = tid & 31;
    const int bm = blockIdx.x * 128;
    const int bn = blockIdx.y * 128;
    const int wm = bm + (w & 1) * 64;
    const int wn = bn + (w >> 1) * 32;

    v8f acc[4][2];
#pragma unroll
    for (int r = 0; r < 4; ++r)
#pragma unroll
        for (int c = 0; c < 2; ++c) acc[r][c] = (v8f){0, 0, 0, 0, 0, 0, 0, 0};

    for (int k0 = 0; k0 < K; k0 += 32) {
        if (k0 + 32 < K) { // keep the next K-slab in flight (global_prefetch_b8)
            __builtin_prefetch(A  + (size_t)(wm + (lane & 15)) * K + k0 + 32, 0, 3);
            __builtin_prefetch(Bt + (size_t)(wn + (lane & 15)) * K + k0 + 32, 0, 3);
        }
        v16bf af[4], bfg[2];
#pragma unroll
        for (int r = 0; r < 4; ++r) af[r] = load_a_frag(A, K, wm + r * 16, k0, lane);
#pragma unroll
        for (int c = 0; c < 2; ++c) bfg[c] = load_b_frag(Bt, K, wn + c * 16, k0, lane);
#pragma unroll
        for (int r = 0; r < 4; ++r)
#pragma unroll
            for (int c = 0; c < 2; ++c) acc[r][c] = wmma_bf16(af[r], bfg[c], acc[r][c]);
    }

    // epilogue: C-frag element j of lane -> row (j + 8*(lane>=16)), col (lane&15)
#pragma unroll
    for (int r = 0; r < 4; ++r) {
#pragma unroll
        for (int c = 0; c < 2; ++c) {
            int colg  = wn + c * 16 + (lane & 15);
            int rbase = wm + r * 16 + ((lane & 16) ? 8 : 0);
#pragma unroll
            for (int j = 0; j < 8; ++j) {
                int m = rbase + j;
                float v = acc[r][c][j];
                switch (mode) {
                case EPI_STORE_BF16:
                    ((bf16*)out0)[(size_t)m * N + colg] = f2bf(v);
                    break;
                case EPI_QPROJ: { // -> [b][n][i][d], two biased copies
                    int i = m >> 3, b = m & 7, n = colg >> 6, d = colg & 63;
                    size_t idx = ((((size_t)b * NH + n) * QLEN) + i) * DH + d;
                    ((bf16*)out0)[idx] = f2bf(v + p0[colg]);
                    ((bf16*)out1)[idx] = f2bf(v + p1[colg]);
                } break;
                case EPI_KPROJ: { // -> [b][n][j][d]
                    int jj = m >> 3, b = m & 7, n = colg >> 6, d = colg & 63;
                    ((bf16*)out0)[((((size_t)b * NH + n) * KLEN) + jj) * DH + d] = f2bf(v);
                } break;
                case EPI_VPROJ: { // -> transposed [b][n][d][j]
                    int jj = m >> 3, b = m & 7, n = colg >> 6, d = colg & 63;
                    ((bf16*)out0)[((((size_t)b * NH + n) * DH) + d) * KLEN + jj] = f2bf(v);
                } break;
                case EPI_RPROJ: { // -> [n][rj][d]
                    int n = colg >> 6, d = colg & 63;
                    ((bf16*)out0)[(((size_t)n * RLEN) + m) * DH + d] = f2bf(v);
                } break;
                case EPI_OPROJ: // residual add: attn_out + h
                    ((float*)out0)[(size_t)m * N + colg] = v + p2[(size_t)m * N + colg];
                    break;
                case EPI_FF1:   // bias + GELU -> bf16
                    ((bf16*)out0)[(size_t)m * N + colg] = f2bf(geluf(v + p0[colg]));
                    break;
                case EPI_FF2:   // bias + residual -> f32
                    ((float*)out0)[(size_t)m * N + colg] =
                        v + p0[colg] + p2[(size_t)m * N + colg];
                    break;
                }
            }
        }
    }
}

// ---------------- fused rel-attention kernel ----------------
// grid: (QLEN/32, NH, BATCH), 256 thr (8 waves), dynamic LDS:
//   sc   f32 [32][1024]  (score tile)                        128 KB
//   pr   bf16[32][1024]  (softmax probs -> WMMA A)            64 KB
//   q1t  bf16[32][64]    (TDM-staged biased q tile #1)         4 KB
//   q2t  bf16[32][64]    (TDM-staged biased q tile #2)         4 KB
#define SC_BYTES  (32 * 1024 * (int)sizeof(float))
#define PR_BYTES  (32 * 1024 * (int)sizeof(bf16))
#define QT_BYTES  (32 * DH * (int)sizeof(bf16))
#define ATTN_SMEM (SC_BYTES + PR_BYTES + 2 * QT_BYTES)

__global__ __launch_bounds__(256) void attn_kernel(
        const bf16* __restrict__ qb1,   // [B][NH][QLEN][DH] (q + r_w_bias)
        const bf16* __restrict__ qb2,   // [B][NH][QLEN][DH] (q + r_r_bias)
        const bf16* __restrict__ kh,    // [B][NH][KLEN][DH]
        const bf16* __restrict__ vT,    // [B][NH][DH][KLEN]
        const bf16* __restrict__ rh,    // [NH][RLEN][DH]
        bf16* __restrict__ av) {        // [QLEN*B][NH*DH]
    extern __shared__ char smem_raw[];
    float* sc  = (float*)smem_raw;
    bf16*  pr  = (bf16*)(smem_raw + SC_BYTES);
    bf16*  q1t = (bf16*)(smem_raw + SC_BYTES + PR_BYTES);
    bf16*  q2t = (bf16*)(smem_raw + SC_BYTES + PR_BYTES + QT_BYTES);

    const int i0 = blockIdx.x * 32;
    const int n  = blockIdx.y;
    const int b  = blockIdx.z;
    const int tid = threadIdx.x, w = tid >> 5, lane = tid & 31;

    const bf16* q1 = qb1 + (((size_t)b * NH + n) * QLEN) * DH;
    const bf16* q2 = qb2 + (((size_t)b * NH + n) * QLEN) * DH;
    const bf16* kp = kh  + (((size_t)b * NH + n) * KLEN) * DH;
    const bf16* vp = vT  + (((size_t)b * NH + n) * DH)   * KLEN;
    const bf16* rp = rh  + ((size_t)n * RLEN) * DH;

#ifdef HAVE_TDM
    // ---- stage both biased q tiles (32 x 64 bf16) into LDS via the TDM ----
    if (tid < 32) {   // one wave issues; TENSORcnt is per-wave
        tdm_load_2d((unsigned)(SC_BYTES + PR_BYTES),
                    q1 + (size_t)i0 * DH, DH, 32, DH);
        tdm_load_2d((unsigned)(SC_BYTES + PR_BYTES + QT_BYTES),
                    q2 + (size_t)i0 * DH, DH, 32, DH);
        wait_tensorcnt0();
    }
    __syncthreads();
    const bf16* q1src = q1t;  const int q1m0 = 0;   const int q1ld = DH;
    const bf16* q2src = q2t;  const int q2m0 = 0;   const int q2ld = DH;
#else
    const bf16* q1src = q1;   const int q1m0 = i0;  const int q1ld = DH;
    const bf16* q2src = q2;   const int q2m0 = i0;  const int q2ld = DH;
#endif

    // ---- phase 1: ac = (q + r_w_bias) @ k^T   -> sc[32][1024]
    v16bf a1[2][2];
#pragma unroll
    for (int rt = 0; rt < 2; ++rt)
#pragma unroll
        for (int ks = 0; ks < 2; ++ks)
            a1[rt][ks] = load_a_frag(q1src, q1ld, q1m0 + rt * 16, ks * 32, lane);

    for (int t = 0; t < 8; ++t) {            // 64 col tiles / 8 waves
        int ct = w * 8 + t;
        v16bf b0 = load_b_frag(kp, DH, ct * 16, 0, lane);
        v16bf b1 = load_b_frag(kp, DH, ct * 16, 32, lane);
#pragma unroll
        for (int rt = 0; rt < 2; ++rt) {
            v8f c8 = (v8f){0, 0, 0, 0, 0, 0, 0, 0};
            c8 = wmma_bf16(a1[rt][0], b0, c8);
            c8 = wmma_bf16(a1[rt][1], b1, c8);
            int col   = ct * 16 + (lane & 15);
            int rbase = rt * 16 + ((lane & 16) ? 8 : 0);
#pragma unroll
            for (int j = 0; j < 8; ++j) sc[(rbase + j) * 1024 + col] = c8[j];
        }
    }
    __syncthreads();

    // ---- phase 2: bd with rel_shift fused: raw[i][rj] -> col (rj - QLEN + i)
    v16bf a2[2][2];
#pragma unroll
    for (int rt = 0; rt < 2; ++rt)
#pragma unroll
        for (int ks = 0; ks < 2; ++ks)
            a2[rt][ks] = load_a_frag(q2src, q2ld, q2m0 + rt * 16, ks * 32, lane);

    for (int t = 0; t < 12; ++t) {           // 96 rel-pos tiles / 8 waves
        int ct = w * 12 + t;
        v16bf b0 = load_b_frag(rp, DH, ct * 16, 0, lane);
        v16bf b1 = load_b_frag(rp, DH, ct * 16, 32, lane);
#pragma unroll
        for (int rt = 0; rt < 2; ++rt) {
            v8f c8 = (v8f){0, 0, 0, 0, 0, 0, 0, 0};
            c8 = wmma_bf16(a2[rt][0], b0, c8);
            c8 = wmma_bf16(a2[rt][1], b1, c8);
            int rj    = ct * 16 + (lane & 15);
            int rbase = rt * 16 + ((lane & 16) ? 8 : 0);
#pragma unroll
            for (int j = 0; j < 8; ++j) {
                int rl  = rbase + j;
                int col = rj - QLEN + (i0 + rl);   // injective per row -> race-free RMW
                if (col >= 0 && col < KLEN) sc[rl * 1024 + col] += c8[j];
            }
        }
    }
    __syncthreads();

    // ---- phase 3: masked softmax over klen (scale 1/sqrt(DH)); wave = 4 rows
    const float scale = 0.125f;
    for (int r = 0; r < 4; ++r) {
        int rl = w * 4 + r;
        int limit = (i0 + rl) + MLEN;        // cols <= limit are valid (causal+mem)
        float mx = -1e30f;
        for (int c = lane; c < KLEN; c += 32) {
            float v = sc[rl * 1024 + c];
            if (c <= limit) mx = fmaxf(mx, v);
        }
#pragma unroll
        for (int off = 16; off; off >>= 1) mx = fmaxf(mx, __shfl_xor(mx, off, 32));
        float sum = 0.f;
        for (int c = lane; c < KLEN; c += 32) {
            float v = sc[rl * 1024 + c];
            float e = (c <= limit) ? __expf(scale * (v - mx)) : 0.f;
            sum += e;
            sc[rl * 1024 + c] = e;
        }
#pragma unroll
        for (int off = 16; off; off >>= 1) sum += __shfl_xor(sum, off, 32);
        float inv = 1.f / sum;
        for (int c = lane; c < KLEN; c += 32)
            pr[rl * 1024 + c] = f2bf(sc[rl * 1024 + c] * inv);
    }
    __syncthreads();

    // ---- phase 4: attn_vec = prob @ v   (A from LDS probs, B from vT)
    int rt = w >> 2, ctn = w & 3;            // 2 row tiles x 4 col tiles
    v8f c8 = (v8f){0, 0, 0, 0, 0, 0, 0, 0};
    for (int ks = 0; ks < KLEN / 32; ++ks) {
        v16bf af = load_a_frag(pr, 1024, rt * 16, ks * 32, lane);
        v16bf bfv = load_b_frag(vp, KLEN, ctn * 16, ks * 32, lane);
        c8 = wmma_bf16(af, bfv, c8);
    }
    {
        int colh  = n * DH + ctn * 16 + (lane & 15);
        int rbase = rt * 16 + ((lane & 16) ? 8 : 0);
#pragma unroll
        for (int j = 0; j < 8; ++j) {
            int i = i0 + rbase + j;
            av[((size_t)i * BATCH + b) * (NH * DH) + colh] = f2bf(c8[j]);
        }
    }
}

// ---------------- host launcher ----------------

static inline size_t align_up256(size_t x) { return (x + 255) & ~(size_t)255; }

extern "C" void kernel_launch(void* const* d_in, const int* in_sizes, int n_in,
                              void* d_out, int out_size, void* d_ws, size_t ws_size,
                              hipStream_t stream) {
    (void)in_sizes; (void)n_in; (void)out_size; (void)ws_size;
    const int*   inp_ids  = (const int*)  d_in[0];
    const float* mems     = (const float*)d_in[1];
    const float* lut      = (const float*)d_in[2];
    const float* q_w      = (const float*)d_in[3];
    const float* k_w      = (const float*)d_in[4];
    const float* v_w      = (const float*)d_in[5];
    const float* r_w      = (const float*)d_in[6];
    const float* o_w      = (const float*)d_in[7];
    const float* r_w_bias = (const float*)d_in[8];
    const float* r_r_bias = (const float*)d_in[9];
    const float* ln1_g    = (const float*)d_in[10];
    const float* ln1_b    = (const float*)d_in[11];
    const float* ff_w1    = (const float*)d_in[12];
    const float* ff_b1    = (const float*)d_in[13];
    const float* ff_w2    = (const float*)d_in[14];
    const float* ff_b2    = (const float*)d_in[15];
    const float* ln2_g    = (const float*)d_in[16];
    const float* ln2_b    = (const float*)d_in[17];

    char* ws = (char*)d_ws;
    size_t off = 0;
    auto alloc = [&](size_t bytes) -> void* {
        void* p = ws + off; off = align_up256(off + bytes); return p;
    };

    bf16*  cat_b  = (bf16*) alloc((size_t)NTOKK * DMODEL * 2);
    float* h_f    = (float*)alloc((size_t)NTOKQ * DMODEL * 4);
    bf16*  pos_b  = (bf16*) alloc((size_t)RLEN * DMODEL * 2);
    bf16*  qw_t   = (bf16*) alloc((size_t)DMODEL * DMODEL * 2);
    bf16*  kw_t   = (bf16*) alloc((size_t)DMODEL * DMODEL * 2);
    bf16*  vw_t   = (bf16*) alloc((size_t)DMODEL * DMODEL * 2);
    bf16*  rw_t   = (bf16*) alloc((size_t)DMODEL * DMODEL * 2);
    bf16*  ow_b   = (bf16*) alloc((size_t)DMODEL * DMODEL * 2);
    bf16*  ffw1_t = (bf16*) alloc((size_t)DMODEL * DI * 2);
    bf16*  ffw2_t = (bf16*) alloc((size_t)DI * DMODEL * 2);
    bf16*  qb1    = (bf16*) alloc((size_t)BATCH * NH * QLEN * DH * 2);
    bf16*  qb2    = (bf16*) alloc((size_t)BATCH * NH * QLEN * DH * 2);
    bf16*  k_head = (bf16*) alloc((size_t)BATCH * NH * KLEN * DH * 2);
    bf16*  vT     = (bf16*) alloc((size_t)BATCH * NH * KLEN * DH * 2);
    bf16*  r_head = (bf16*) alloc((size_t)NH * RLEN * DH * 2);
    bf16*  av     = (bf16*) alloc((size_t)NTOKQ * DMODEL * 2);
    float* res1   = (float*)alloc((size_t)NTOKQ * DMODEL * 4);
    float* out1_f = (float*)alloc((size_t)NTOKQ * DMODEL * 4);
    bf16*  out1_b = (bf16*) alloc((size_t)NTOKQ * DMODEL * 2);
    bf16*  ffa    = (bf16*) alloc((size_t)NTOKQ * DI * 2);
    float* res2   = (float*)alloc((size_t)NTOKQ * DMODEL * 4);

    dim3 blk(256);
    const int GS = 2048; // grid-stride helper grid

    // weight conversions (f32 -> bf16, pre-transposed to [N][K])
    convert_transpose_kernel<<<GS, blk, 0, stream>>>(q_w, qw_t, DMODEL, DMODEL);
    convert_transpose_kernel<<<GS, blk, 0, stream>>>(k_w, kw_t, DMODEL, DMODEL);
    convert_transpose_kernel<<<GS, blk, 0, stream>>>(v_w, vw_t, DMODEL, DMODEL);
    convert_transpose_kernel<<<GS, blk, 0, stream>>>(r_w, rw_t, DMODEL, DMODEL);
    convert_copy_kernel     <<<GS, blk, 0, stream>>>(o_w, ow_b, (size_t)DMODEL * DMODEL);
    convert_transpose_kernel<<<GS, blk, 0, stream>>>(ff_w1, ffw1_t, DMODEL, DI);
    convert_transpose_kernel<<<GS, blk, 0, stream>>>(ff_w2, ffw2_t, DI, DMODEL);

    embed_kernel <<<GS, blk, 0, stream>>>(inp_ids, mems, lut, cat_b, h_f);
    posemb_kernel<<<GS, blk, 0, stream>>>(pos_b);

    const bf16* h_b = cat_b + (size_t)MLEN * BATCH * DMODEL; // query rows of cat

    // projections (all bf16 WMMA)
    gemm_bf16_wmma_kernel<<<dim3(NTOKQ / 128, DMODEL / 128), blk, 0, stream>>>(
        h_b, qw_t, NTOKQ, DMODEL, DMODEL, EPI_QPROJ, r_w_bias, r_r_bias, nullptr, qb1, qb2);
    gemm_bf16_wmma_kernel<<<dim3(NTOKK / 128, DMODEL / 128), blk, 0, stream>>>(
        cat_b, kw_t, NTOKK, DMODEL, DMODEL, EPI_KPROJ, nullptr, nullptr, nullptr, k_head, nullptr);
    gemm_bf16_wmma_kernel<<<dim3(NTOKK / 128, DMODEL / 128), blk, 0, stream>>>(
        cat_b, vw_t, NTOKK, DMODEL, DMODEL, EPI_VPROJ, nullptr, nullptr, nullptr, vT, nullptr);
    gemm_bf16_wmma_kernel<<<dim3(RLEN / 128, DMODEL / 128), blk, 0, stream>>>(
        pos_b, rw_t, RLEN, DMODEL, DMODEL, EPI_RPROJ, nullptr, nullptr, nullptr, r_head, nullptr);

    // fused rel-attention: scores + rel_shift + softmax + AV (TDM-staged q tiles)
    attn_kernel<<<dim3(QLEN / 32, NH, BATCH), blk, ATTN_SMEM, stream>>>(
        qb1, qb2, k_head, vT, r_head, av);

    // output projection + residual
    gemm_bf16_wmma_kernel<<<dim3(NTOKQ / 128, DMODEL / 128), blk, 0, stream>>>(
        av, ow_b, NTOKQ, DMODEL, DMODEL, EPI_OPROJ, nullptr, nullptr, h_f, res1, nullptr);
    layernorm_kernel<<<NTOKQ, blk, 0, stream>>>(res1, ln1_g, ln1_b, out1_f, out1_b);

    // FFN
    gemm_bf16_wmma_kernel<<<dim3(NTOKQ / 128, DI / 128), blk, 0, stream>>>(
        out1_b, ffw1_t, NTOKQ, DI, DMODEL, EPI_FF1, ff_b1, nullptr, nullptr, ffa, nullptr);
    gemm_bf16_wmma_kernel<<<dim3(NTOKQ / 128, DMODEL / 128), blk, 0, stream>>>(
        ffa, ffw2_t, NTOKQ, DMODEL, DI, EPI_FF2, ff_b2, nullptr, out1_f, res2, nullptr);

    layernorm_kernel<<<NTOKQ, blk, 0, stream>>>(res2, ln2_g, ln2_b, (float*)d_out, nullptr);
}
